// Seq2SeqRNN_39565238731407
// MI455X (gfx1250) — matrix operand order repaired and checked by reading, not compiled, not run on hardware
//
#include <hip/hip_runtime.h>
#include <hip/hip_bf16.h>

// ---------------------------------------------------------------------------
// Seq2Seq GRU + attention for MI455X (gfx1250, wave32, WMMA bf16, TDM)
//   * weights f32 -> bf16 once (fc_W bf16 = 82MB, L2-resident at 192MB)
//   * sequential recurrence with fused dual-GEMM WMMA gate kernels
//   * output projection hoisted out of the scan: ONE batched WMMA GEMM
//     [2016x1280]x[1280x32000], 2x4 register blocking, A-panel staged into
//     LDS per block via TENSOR_LOAD_TO_LDS (TENSORcnt) and shared by 4 waves
// ---------------------------------------------------------------------------

#define B_   32
#define SS_  64
#define ST_  64
#define E_   256
#define H_   512
#define V_   32000
#define H3_  1536      // 3*H
#define KD0_ 768       // E+H  (decoder layer-0 input)
#define KP_  1280      // 2H+E (pred_in width)
#define MR_  2016      // (ST-1)*B rows of pred_in

typedef __attribute__((ext_vector_type(16))) __bf16 v16bf;
typedef __attribute__((ext_vector_type(8)))  float  v8f;
typedef __attribute__((ext_vector_type(4)))  unsigned int v4u;
typedef __attribute__((ext_vector_type(8)))  int v8i;
typedef __attribute__((ext_vector_type(4)))  int v4i;

#ifndef __has_builtin
#define __has_builtin(x) 0
#endif
#if __has_builtin(__builtin_amdgcn_tensor_load_to_lds) && \
    __has_builtin(__builtin_amdgcn_s_wait_tensorcnt)
#define HAVE_TDM 1
#else
#define HAVE_TDM 0
#endif

__device__ __forceinline__ unsigned short f2bf(float f) {
  unsigned int u = __float_as_uint(f);
  u += 0x7FFFu + ((u >> 16) & 1u);   // round-to-nearest-even
  return (unsigned short)(u >> 16);
}

__device__ __forceinline__ float sigmoid_(float x) {
  return 1.0f / (1.0f + __expf(-x));
}

// A fragment: 16x32 bf16, row-major source (lda in elements).
// Layout used here: lanes 0-15 row M=l, half selects K 0..15 vs 16..31, contiguous K-pairs
__device__ __forceinline__ v16bf load_a_frag(const unsigned short* __restrict__ A,
                                             int lda, int k0) {
  const int lane = threadIdx.x & 31;
  const int l = lane & 15, half = lane >> 4;
  const unsigned short* row = A + l * lda + k0 + half * 16;
  union { v16bf v; unsigned int u[8]; } f;
#pragma unroll
  for (int i = 0; i < 8; ++i) {
    f.u[i]     = *(const unsigned int*)(row + 2 * i);
  }
  return f.v;
}

// B fragment: 32x16 bf16 from weight stored [N,K] row-major (i.e. B^T),
// lane l = column n, half selects K 0..15 vs 16..31, contiguous K-pairs.
__device__ __forceinline__ v16bf load_b_frag(const unsigned short* __restrict__ W,
                                             int ldw, int k0) {
  const int lane = threadIdx.x & 31;
  const int l = lane & 15, half = lane >> 4;
  const unsigned short* row = W + l * ldw + k0 + half * 16;
  union { v16bf v; unsigned int u[8]; } f;
#pragma unroll
  for (int i = 0; i < 8; ++i)
    f.u[i] = *(const unsigned int*)(row + 2 * i);
  return f.v;
}

#define WMMA_BF16(ACC, AF, BF) \
  __builtin_amdgcn_wmma_f32_16x16x32_bf16(false, (AF), false, (BF), (short)0, (ACC), false, false)

// ---------------------------------------------------------------------------
// utility kernels
// ---------------------------------------------------------------------------
__global__ void k_f2bf(const float* __restrict__ s, unsigned short* __restrict__ d, int n) {
  int i = blockIdx.x * blockDim.x + threadIdx.x;
  if (i < n) d[i] = f2bf(s[i]);
}

__global__ void k_zero32(unsigned int* __restrict__ p, int n) {
  int i = blockIdx.x * blockDim.x + threadIdx.x;
  if (i < n) p[i] = 0u;
}

__global__ void k_zero_t0(float* __restrict__ out) {
  int i = blockIdx.x * blockDim.x + threadIdx.x;
  if (i >= B_ * V_) return;
  int b = i / V_, v = i % V_;
  out[(size_t)b * ST_ * V_ + v] = 0.0f;
}

// Embedding gather -> bf16, laid out [T][B][ld], optionally duplicated
__global__ void k_embed(const int* __restrict__ tok, int tok_ld,
                        const float* __restrict__ emb, int T,
                        unsigned short* __restrict__ dst0, int ld0,
                        unsigned short* __restrict__ dst1, int ld1) {
  int i = blockIdx.x * blockDim.x + threadIdx.x;
  if (i >= T * B_ * E_) return;
  int e = i % E_;
  int b = (i / E_) & (B_ - 1);
  int t = i / (E_ * B_);
  int id = tok[b * tok_ld + t];
  unsigned short bf = f2bf(emb[(long)id * E_ + e]);
  dst0[(size_t)(t * B_ + b) * ld0 + e] = bf;
  if (dst1) dst1[(size_t)(t * B_ + b) * ld1 + e] = bf;
}

// ---------------------------------------------------------------------------
// generic WMMA GEMM: C[M,N] = A[M,K] * W[N,K]^T (+bias), one 16x16 tile/wave
// ---------------------------------------------------------------------------
__global__ void k_gemm(const unsigned short* __restrict__ A, int lda,
                       const unsigned short* __restrict__ W, int ldw,
                       float* __restrict__ C, int ldc,
                       int mtiles, int ntiles, int K,
                       const float* __restrict__ bias) {
  int wid = blockIdx.x * (blockDim.x >> 5) + (threadIdx.x >> 5);
  if (wid >= mtiles * ntiles) return;
  int mt = wid % mtiles, nt = wid / mtiles;
  const unsigned short* Ab = A + (size_t)mt * 16 * lda;
  const unsigned short* Wb = W + (size_t)nt * 16 * ldw;
  v8f acc = {};
  for (int k0 = 0; k0 < K; k0 += 32)
    acc = WMMA_BF16(acc, load_a_frag(Ab, lda, k0), load_b_frag(Wb, ldw, k0));
  int lane = threadIdx.x & 31, l = lane & 15, half = lane >> 4;
  int n = nt * 16 + l;
  float bv = bias ? bias[n] : 0.0f;
#pragma unroll
  for (int r = 0; r < 8; ++r) {
    int m = mt * 16 + half * 8 + r;
    C[(size_t)m * ldc + n] = acc[r] + bv;
  }
}

// ---------------------------------------------------------------------------
// fused GRU gate GEMMs: Gi = X*Wih^T + bih ; Gh = Hprev*Whh^T + bhh
// ---------------------------------------------------------------------------
__global__ void k_gru_gates(const unsigned short* __restrict__ X, int ldx, int Kx,
                            const unsigned short* __restrict__ Hb,
                            const unsigned short* __restrict__ Wih,
                            const unsigned short* __restrict__ Whh,
                            const float* __restrict__ bih,
                            const float* __restrict__ bhh,
                            float* __restrict__ Gi, float* __restrict__ Gh) {
  int wid = blockIdx.x * (blockDim.x >> 5) + (threadIdx.x >> 5);
  if (wid >= 2 * (H3_ / 16)) return;
  int mt = wid & 1, nt = wid >> 1;
  const unsigned short* Xb = X  + (size_t)mt * 16 * ldx;
  const unsigned short* Hh = Hb + (size_t)mt * 16 * H_;
  const unsigned short* Wi = Wih + (size_t)nt * 16 * Kx;
  const unsigned short* Wh = Whh + (size_t)nt * 16 * H_;
  v8f ai = {}, ah = {};
  for (int k0 = 0; k0 < Kx; k0 += 32)
    ai = WMMA_BF16(ai, load_a_frag(Xb, ldx, k0), load_b_frag(Wi, Kx, k0));
  for (int k0 = 0; k0 < H_; k0 += 32)
    ah = WMMA_BF16(ah, load_a_frag(Hh, H_, k0), load_b_frag(Wh, H_, k0));
  int lane = threadIdx.x & 31, l = lane & 15, half = lane >> 4;
  int n = nt * 16 + l;
  float bi = bih[n], bh = bhh[n];
#pragma unroll
  for (int r = 0; r < 8; ++r) {
    int m = mt * 16 + half * 8 + r;
    Gi[(size_t)m * H3_ + n] = ai[r] + bi;
    Gh[(size_t)m * H3_ + n] = ah[r] + bh;
  }
}

// GRU elementwise combine: h' = (1-z)*n + z*h
__global__ void k_gru_combine(const float* __restrict__ Gi, const float* __restrict__ Gh,
                              float* __restrict__ h, unsigned short* __restrict__ hb,
                              float* __restrict__ cf, int ld_cf,
                              unsigned short* __restrict__ cb, int ld_cb) {
  int i = blockIdx.x * blockDim.x + threadIdx.x;
  if (i >= B_ * H_) return;
  int b = i >> 9, hh = i & (H_ - 1);
  size_t base = (size_t)b * H3_;
  float r = sigmoid_(Gi[base + hh]        + Gh[base + hh]);
  float z = sigmoid_(Gi[base + H_ + hh]   + Gh[base + H_ + hh]);
  float n = tanhf  (Gi[base + 2*H_ + hh] + r * Gh[base + 2*H_ + hh]);
  float hn = (1.0f - z) * n + z * h[i];
  h[i] = hn;
  unsigned short bf = f2bf(hn);
  hb[i] = bf;
  if (cf) cf[(size_t)b * ld_cf + hh] = hn;
  if (cb) cb[(size_t)b * ld_cb + hh] = bf;
}

// attention scores: one wave per (s,b); score = v . tanh(enc_proj + dec_proj)
__global__ void k_scores(const float* __restrict__ enc_proj,
                         const float* __restrict__ dec_proj,
                         const float* __restrict__ av,
                         float* __restrict__ scores) {
  int wid = blockIdx.x * (blockDim.x >> 5) + (threadIdx.x >> 5);
  if (wid >= SS_ * B_) return;
  int lane = threadIdx.x & 31;
  int b = wid & (B_ - 1);
  const float* ep = enc_proj + (size_t)wid * H_;   // row s*B+b
  const float* dp = dec_proj + (size_t)b * H_;
  float sum = 0.0f;
  for (int hh = lane; hh < H_; hh += 32)
    sum += av[hh] * tanhf(ep[hh] + dp[hh]);
#pragma unroll
  for (int off = 16; off; off >>= 1) sum += __shfl_down(sum, off, 32);
  if (lane == 0) scores[b * SS_ + (wid >> 5)] = sum;
}

// softmax over SS=64 per batch row; one wave per row (2 elems/lane)
__global__ void k_softmax(const float* __restrict__ scores, float* __restrict__ w) {
  int b = blockIdx.x, lane = threadIdx.x;
  float x0 = scores[b * SS_ + lane], x1 = scores[b * SS_ + 32 + lane];
  float m = fmaxf(x0, x1);
#pragma unroll
  for (int off = 16; off; off >>= 1) m = fmaxf(m, __shfl_xor(m, off, 32));
  float e0 = __expf(x0 - m), e1 = __expf(x1 - m);
  float s = e0 + e1;
#pragma unroll
  for (int off = 16; off; off >>= 1) s += __shfl_xor(s, off, 32);
  w[b * SS_ + lane] = e0 / s;
  w[b * SS_ + 32 + lane] = e1 / s;
}

// context: ctx[b][h] = sum_s w[b][s] * enc_out[s][b][h]
__global__ void k_ctx(const float* __restrict__ w,
                      const float* __restrict__ enc_out_f,
                      unsigned short* __restrict__ rnn_in_t,
                      unsigned short* __restrict__ pred_in_t) {
  int i = blockIdx.x * blockDim.x + threadIdx.x;
  if (i >= B_ * H_) return;
  int b = i >> 9, hh = i & (H_ - 1);
  float acc = 0.0f;
#pragma unroll 4
  for (int s = 0; s < SS_; ++s)
    acc += w[b * SS_ + s] * enc_out_f[((size_t)s * B_ + b) * H_ + hh];
  unsigned short bf = f2bf(acc);
  rnn_in_t[(size_t)b * KD0_ + E_ + hh] = bf;
  pred_in_t[(size_t)b * KP_ + H_ + hh] = bf;
}

// ---------------------------------------------------------------------------
// batched output projection: [2016x1280] x fc_W[32000x1280]^T
// block = 128 threads (4 waves), block tile = 32 rows x 256 cols
// A panel (32x1280 bf16 = 80KB) staged into LDS via TENSOR_LOAD_TO_LDS and
// shared by all 4 waves; each wave: 2 M-tiles x 4 N-tiles register blocking.
// Output scattered into out[b][t+1][v] with fused bias. (t == row-panel id)
// ---------------------------------------------------------------------------
__global__ void k_fc(const unsigned short* __restrict__ P,
                     const unsigned short* __restrict__ Wf,
                     const float* __restrict__ bias,
                     float* __restrict__ out) {
  extern __shared__ unsigned short ldsA[];         // [32][KP_] bf16 = 80KB
  const int mt2 = blockIdx.x % (MR_ / 32);         // 0..62  -> also t index
  const int ngg = blockIdx.x / (MR_ / 32);         // 0..124 (256-col group)
  const unsigned short* Ag = P + (size_t)mt2 * 32 * KP_;

#if HAVE_TDM
  if ((threadIdx.x >> 5) == 0) {
    // Tensor DMA descriptor (ISA cdna5 ch.8): 2D tile, 2-byte elements,
    // tile 1280x32, row stride 1280, dest LDS offset of ldsA.
    unsigned long long ga = (unsigned long long)(size_t)Ag;
    unsigned lds_off = (unsigned)(size_t)(void*)&ldsA[0];
    v4u g0;
    g0[0] = 1u;                                          // count=1, user D#
    g0[1] = lds_off;                                     // lds_addr (bytes)
    g0[2] = (unsigned)(ga & 0xFFFFFFFFu);                // global_addr[31:0]
    g0[3] = (unsigned)((ga >> 32) & 0x01FFFFFFu) | (2u << 30);  // [56:32] | type=2
    v8i g1;
    g1[0] = 0x00010000;                                  // data_size=1 (2B)
    g1[1] = (int)((KP_ & 0xFFFF) << 16);                 // tensor_dim0[15:0]
    g1[2] = (int)((KP_ >> 16) | (32u << 16));            // dim0[31:16] | tensor_dim1[15:0]=32
    g1[3] = (int)(((unsigned)KP_) << 16);                // dim1[31:16]=0 | tile_dim0=1280
    g1[4] = 32;                                          // tile_dim1=32, tile_dim2=0
    g1[5] = KP_;                                         // tensor_dim0_stride[31:0]
    g1[6] = 0;                                           // stride[47:32], dim1_stride lo
    g1[7] = 0;
    v4i g2 = {0, 0, 0, 0};                               // group2: 2D -> unused dims
    v4i g3 = {0, 0, 0, 0};                               // group3: unused
    v8i g4 = {0, 0, 0, 0, 0, 0, 0, 0};                   // extra group (clang-23 form)
    __builtin_amdgcn_tensor_load_to_lds(g0, g1, g2, g3, g4, 0);
    __builtin_amdgcn_s_wait_tensorcnt(0);
  }
#else
  // fallback: cooperative b128 global->LDS copy
  for (int i = threadIdx.x; i < 32 * KP_ / 8; i += blockDim.x)
    ((uint4*)ldsA)[i] = ((const uint4*)Ag)[i];
#endif
  __syncthreads();

  const int w = threadIdx.x >> 5;
  const unsigned short* W0 = Wf + (size_t)(ngg * 256 + w * 64 +  0) * KP_;
  const unsigned short* W1 = Wf + (size_t)(ngg * 256 + w * 64 + 16) * KP_;
  const unsigned short* W2 = Wf + (size_t)(ngg * 256 + w * 64 + 32) * KP_;
  const unsigned short* W3 = Wf + (size_t)(ngg * 256 + w * 64 + 48) * KP_;
  const unsigned short* A0 = &ldsA[0];
  const unsigned short* A1 = &ldsA[16 * KP_];

  v8f c00 = {}, c01 = {}, c02 = {}, c03 = {};
  v8f c10 = {}, c11 = {}, c12 = {}, c13 = {};
  for (int k0 = 0; k0 < KP_; k0 += 32) {
    __builtin_prefetch(W0 + k0 + 64, 0, 0);       // gfx1250 global_prefetch_b8
    v16bf a0 = load_a_frag(A0, KP_, k0);          // LDS (ds_load) A rows 0-15
    v16bf a1 = load_a_frag(A1, KP_, k0);          // LDS A rows 16-31
    v16bf b0 = load_b_frag(W0, KP_, k0);
    c00 = WMMA_BF16(c00, a0, b0);  c10 = WMMA_BF16(c10, a1, b0);
    v16bf b1 = load_b_frag(W1, KP_, k0);
    c01 = WMMA_BF16(c01, a0, b1);  c11 = WMMA_BF16(c11, a1, b1);
    v16bf b2 = load_b_frag(W2, KP_, k0);
    c02 = WMMA_BF16(c02, a0, b2);  c12 = WMMA_BF16(c12, a1, b2);
    v16bf b3 = load_b_frag(W3, KP_, k0);
    c03 = WMMA_BF16(c03, a0, b3);  c13 = WMMA_BF16(c13, a1, b3);
  }

  const int lane = threadIdx.x & 31, l = lane & 15, half = lane >> 4;
  const int t = mt2;
#define FC_STORE(ACC, MLOC, J)                                                \
  {                                                                           \
    int n = ngg * 256 + w * 64 + (J) * 16 + l;                                \
    float bv = bias[n];                                                       \
    _Pragma("unroll") for (int r = 0; r < 8; ++r) {                           \
      int b = (MLOC) * 16 + half * 8 + r;                                     \
      out[(size_t)b * ST_ * V_ + (size_t)(t + 1) * V_ + n] = ACC[r] + bv;     \
    }                                                                         \
  }
  FC_STORE(c00, 0, 0) FC_STORE(c01, 0, 1) FC_STORE(c02, 0, 2) FC_STORE(c03, 0, 3)
  FC_STORE(c10, 1, 0) FC_STORE(c11, 1, 1) FC_STORE(c12, 1, 2) FC_STORE(c13, 1, 3)
#undef FC_STORE
}

// ---------------------------------------------------------------------------
// host orchestration
// ---------------------------------------------------------------------------
extern "C" void kernel_launch(void* const* d_in, const int* in_sizes, int n_in,
                              void* d_out, int out_size, void* d_ws, size_t ws_size,
                              hipStream_t stream) {
  (void)in_sizes; (void)n_in; (void)out_size; (void)ws_size;
  const int*   src      = (const int*)  d_in[0];
  const int*   tgt      = (const int*)  d_in[1];
  const float* emb_enc  = (const float*)d_in[2];
  const float* emb_dec  = (const float*)d_in[3];
  const float* eWih0    = (const float*)d_in[4];
  const float* eWhh0    = (const float*)d_in[5];
  const float* ebih0    = (const float*)d_in[6];
  const float* ebhh0    = (const float*)d_in[7];
  const float* eWih1    = (const float*)d_in[8];
  const float* eWhh1    = (const float*)d_in[9];
  const float* ebih1    = (const float*)d_in[10];
  const float* ebhh1    = (const float*)d_in[11];
  const float* dWih0    = (const float*)d_in[12];
  const float* dWhh0    = (const float*)d_in[13];
  const float* dbih0    = (const float*)d_in[14];
  const float* dbhh0    = (const float*)d_in[15];
  const float* dWih1    = (const float*)d_in[16];
  const float* dWhh1    = (const float*)d_in[17];
  const float* dbih1    = (const float*)d_in[18];
  const float* dbhh1    = (const float*)d_in[19];
  const float* attn_W1  = (const float*)d_in[20];
  const float* attn_W2  = (const float*)d_in[21];
  const float* attn_v   = (const float*)d_in[22];
  const float* fc_W     = (const float*)d_in[23];
  const float* fc_b     = (const float*)d_in[24];
  float* out = (float*)d_out;

  // workspace bump allocator (256B aligned)
  char* wsp = (char*)d_ws;
  auto alloc = [&](size_t bytes) -> char* {
    char* p = wsp;
    wsp += (bytes + 255) & ~(size_t)255;
    return p;
  };
  typedef unsigned short u16;
  u16* wb_eWih0 = (u16*)alloc((size_t)H3_ * E_  * 2);
  u16* wb_eWhh0 = (u16*)alloc((size_t)H3_ * H_  * 2);
  u16* wb_eWih1 = (u16*)alloc((size_t)H3_ * H_  * 2);
  u16* wb_eWhh1 = (u16*)alloc((size_t)H3_ * H_  * 2);
  u16* wb_dWih0 = (u16*)alloc((size_t)H3_ * KD0_* 2);
  u16* wb_dWhh0 = (u16*)alloc((size_t)H3_ * H_  * 2);
  u16* wb_dWih1 = (u16*)alloc((size_t)H3_ * H_  * 2);
  u16* wb_dWhh1 = (u16*)alloc((size_t)H3_ * H_  * 2);
  u16* wb_W1    = (u16*)alloc((size_t)H_  * H_  * 2);
  u16* wb_W2    = (u16*)alloc((size_t)H_  * H_  * 2);
  u16* wb_fc    = (u16*)alloc((size_t)V_  * KP_ * 2);
  u16* x_enc    = (u16*)alloc((size_t)SS_ * B_ * E_ * 2);
  u16* enc_ob   = (u16*)alloc((size_t)SS_ * B_ * H_ * 2);
  float* enc_of = (float*)alloc((size_t)SS_ * B_ * H_ * 4);
  float* enc_pj = (float*)alloc((size_t)SS_ * B_ * H_ * 4);
  u16* rnn_in   = (u16*)alloc((size_t)(ST_-1) * B_ * KD0_ * 2);
  u16* pred_in  = (u16*)alloc((size_t)(ST_-1) * B_ * KP_  * 2);
  float* h0_f   = (float*)alloc((size_t)B_ * H_ * 4);
  float* h1_f   = (float*)alloc((size_t)B_ * H_ * 4);
  u16* h0_b     = (u16*)alloc((size_t)B_ * H_ * 2);
  u16* h1_b     = (u16*)alloc((size_t)B_ * H_ * 2);
  float* Gi     = (float*)alloc((size_t)B_ * H3_ * 4);
  float* Gh     = (float*)alloc((size_t)B_ * H3_ * 4);
  float* dec_pj = (float*)alloc((size_t)B_ * H_ * 4);
  float* scores = (float*)alloc((size_t)B_ * SS_ * 4);
  float* attw   = (float*)alloc((size_t)B_ * SS_ * 4);

  // ---- weight conversion f32 -> bf16 ----
  const float* csrc[11] = {eWih0, eWhh0, eWih1, eWhh1, dWih0, dWhh0, dWih1,
                           dWhh1, attn_W1, attn_W2, fc_W};
  u16* cdst[11] = {wb_eWih0, wb_eWhh0, wb_eWih1, wb_eWhh1, wb_dWih0, wb_dWhh0,
                   wb_dWih1, wb_dWhh1, wb_W1, wb_W2, wb_fc};
  int cn[11] = {H3_*E_, H3_*H_, H3_*H_, H3_*H_, H3_*KD0_, H3_*H_, H3_*H_,
                H3_*H_, H_*H_, H_*H_, V_*KP_};
  for (int i = 0; i < 11; ++i)
    k_f2bf<<<(cn[i] + 255) / 256, 256, 0, stream>>>(csrc[i], cdst[i], cn[i]);

  // ---- zero initial hidden states ----
  k_zero32<<<(B_*H_ + 255)/256, 256, 0, stream>>>((unsigned int*)h0_f, B_*H_);
  k_zero32<<<(B_*H_ + 255)/256, 256, 0, stream>>>((unsigned int*)h1_f, B_*H_);
  k_zero32<<<(B_*H_/2 + 255)/256, 256, 0, stream>>>((unsigned int*)h0_b, B_*H_/2);
  k_zero32<<<(B_*H_/2 + 255)/256, 256, 0, stream>>>((unsigned int*)h1_b, B_*H_/2);

  // ---- embeddings ----
  k_embed<<<(SS_*B_*E_ + 255)/256, 256, 0, stream>>>(
      src, SS_, emb_enc, SS_, x_enc, E_, (u16*)nullptr, 0);
  k_embed<<<((ST_-1)*B_*E_ + 255)/256, 256, 0, stream>>>(
      tgt, ST_, emb_dec, ST_-1, rnn_in, KD0_, pred_in + (2*H_), KP_);

  const int TPB = 128;
  auto wblk = [](int waves) { return (waves + 3) / 4; };
  const int comb_blk = (B_*H_ + 255) / 256;

  // ---- encoder scan ----
  for (int t = 0; t < SS_; ++t) {
    k_gru_gates<<<wblk(2*(H3_/16)), TPB, 0, stream>>>(
        x_enc + (size_t)t*B_*E_, E_, E_, h0_b, wb_eWih0, wb_eWhh0,
        ebih0, ebhh0, Gi, Gh);
    k_gru_combine<<<comb_blk, 256, 0, stream>>>(
        Gi, Gh, h0_f, h0_b, nullptr, 0, nullptr, 0);
    k_gru_gates<<<wblk(2*(H3_/16)), TPB, 0, stream>>>(
        h0_b, H_, H_, h1_b, wb_eWih1, wb_eWhh1, ebih1, ebhh1, Gi, Gh);
    k_gru_combine<<<comb_blk, 256, 0, stream>>>(
        Gi, Gh, h1_f, h1_b,
        enc_of + (size_t)t*B_*H_, H_, enc_ob + (size_t)t*B_*H_, H_);
  }

  // ---- enc_proj = enc_out @ W1^T  [2048 x 512] ----
  k_gemm<<<wblk((SS_*B_/16)*(H_/16)), TPB, 0, stream>>>(
      enc_ob, H_, wb_W1, H_, enc_pj, H_, SS_*B_/16, H_/16, H_, nullptr);

  // ---- decoder scan (recurrence + attention; fc hoisted out) ----
  for (int t = 0; t < ST_ - 1; ++t) {
    k_gemm<<<wblk(2*(H_/16)), TPB, 0, stream>>>(
        h1_b, H_, wb_W2, H_, dec_pj, H_, 2, H_/16, H_, nullptr);
    k_scores<<<wblk(SS_*B_), TPB, 0, stream>>>(enc_pj, dec_pj, attn_v, scores);
    k_softmax<<<B_, 32, 0, stream>>>(scores, attw);
    k_ctx<<<comb_blk, 256, 0, stream>>>(
        attw, enc_of, rnn_in + (size_t)t*B_*KD0_, pred_in + (size_t)t*B_*KP_);
    k_gru_gates<<<wblk(2*(H3_/16)), TPB, 0, stream>>>(
        rnn_in + (size_t)t*B_*KD0_, KD0_, KD0_, h0_b, wb_dWih0, wb_dWhh0,
        dbih0, dbhh0, Gi, Gh);
    k_gru_combine<<<comb_blk, 256, 0, stream>>>(
        Gi, Gh, h0_f, h0_b, nullptr, 0, nullptr, 0);
    k_gru_gates<<<wblk(2*(H3_/16)), TPB, 0, stream>>>(
        h0_b, H_, H_, h1_b, wb_dWih1, wb_dWhh1, dbih1, dbhh1, Gi, Gh);
    k_gru_combine<<<comb_blk, 256, 0, stream>>>(
        Gi, Gh, h1_f, h1_b, nullptr, 0,
        pred_in + (size_t)t*B_*KP_, KP_);   // c1 -> pred_in cols [0,H)
  }

  // ---- batched output projection (block: 32 rows x 256 cols, LDS A panel) ----
  k_fc<<<(MR_/32) * (V_/256), TPB, 32 * KP_ * 2, stream>>>(pred_in, wb_fc, fc_b, out);
  k_zero_t0<<<(B_*V_ + 255)/256, 256, 0, stream>>>(out);
}
